// Encoder1_84258668413487
// MI455X (gfx1250) — matrix operand order repaired and checked by reading, not compiled, not run on hardware
//
#include <hip/hip_runtime.h>
#include <hip/hip_bf16.h>
#include <stdint.h>

typedef __attribute__((ext_vector_type(16))) _Float16 v16h;
typedef __attribute__((ext_vector_type(8)))  _Float16 v8h;
typedef __attribute__((ext_vector_type(8)))  float    v8f;

#define NN     50000
#define EE     400000
#define F_INC  16
#define SLOPE  0.2f

// ---------------------------------------------------------------------------
// helpers
// ---------------------------------------------------------------------------
__device__ __forceinline__ v8f wmma16(v16h a, v16h b, v8f c) {
  // D = A(16x32 f16) * B(32x16 f16) + C(16x16 f32)
  return __builtin_amdgcn_wmma_f32_16x16x32_f16(false, a, false, b, (short)0, c, false, false);
}

__device__ __forceinline__ v8f splat8(float x) {
  v8f c;
#pragma unroll
  for (int i = 0; i < 8; i++) c[i] = x;
  return c;
}

// B-fragment: pre-packed [frag][lane][16 halves] -> two 16B loads
__device__ __forceinline__ v16h load_frag16(const _Float16* p) {
  union { v16h v; v8h h[2]; } u;
  u.h[0] = *(const v8h*)(p);
  u.h[1] = *(const v8h*)(p + 8);
  return u.v;
}

// A-fragment from row-major f16 activation tile [16][K] in LDS.
// elem j (j<8):  k = 32*kg +      hi*8 + j      (contiguous 16B)
// elem j (j>=8): k = 32*kg + 16 + hi*8 + (j-8)  (contiguous 16B)
__device__ __forceinline__ v16h load_act_frag(const _Float16* act, int K, int kg, int lo, int hi) {
  const _Float16* p = act + lo * K + kg * 32 + hi * 8;
  union { v16h v; v8h h[2]; } u;
  u.h[0] = *(const v8h*)(p);
  u.h[1] = *(const v8h*)(p + 16);
  return u.v;
}

__device__ __forceinline__ void atomicMaxF(float* addr, float v) {
  if (v >= 0.f) atomicMax((int*)addr, __float_as_int(v));
  else          atomicMin((unsigned int*)addr, __float_as_uint(v));
}

__device__ __forceinline__ float leaky(float a) { return a > 0.f ? a : SLOPE * a; }

// ---------------------------------------------------------------------------
// Kernel 0: pack f32 weights [K][Nout] into WMMA B-fragments (f16).
// Fragment f = kg*T + t, layout [f][lane][j]; element (k,n): lane = n + 16*((k/8)%2),
// j = 8*((k%32)/16) + k%8 within the 32-deep K-group.
// ---------------------------------------------------------------------------
__global__ void prep_frags_kernel(const float* __restrict__ W, _Float16* __restrict__ out,
                                  int K, int Nout, int KG) {
  const int T = Nout >> 4;
  const int total = KG * T * 512;
  for (int tid = blockIdx.x * blockDim.x + threadIdx.x; tid < total; tid += gridDim.x * blockDim.x) {
    int f = tid >> 9;
    int r = tid & 511;
    int lane = r >> 4;
    int j = r & 15;
    int kg = f / T, t = f - kg * T;
    int n = lane & 15;
    int k = 32 * kg + ((j >> 3) << 4) + ((lane >> 4) << 3) + (j & 7);
    float v = (k < K) ? W[(size_t)k * Nout + 16 * t + n] : 0.f;
    out[tid] = (_Float16)v;
  }
}

// ---------------------------------------------------------------------------
// Kernel 1: fused EdgeMLP (3->128->64->512) + NNConv message + scatter-add.
// One wave = 16 edges. 88 v_wmma per tile. Weights staged in LDS.
// 8 waves / block; LDS halves: W1f[4096] | W2f[8192] | W3f[32768] | per-wave act
// ---------------------------------------------------------------------------
#define WAVES    8
#define LDS_W2F  4096
#define LDS_W3F  12288
#define LDS_ACT  45056
#define LDS_HALVES (LDS_ACT + WAVES * 3072)   // 69632 halves = 139264 bytes

__global__ __launch_bounds__(WAVES * 32)
__attribute__((amdgpu_waves_per_eu(2)))
void nnconv_edge_kernel(
    const float* __restrict__ x, const float* __restrict__ edge_attr,
    const int* __restrict__ src, const int* __restrict__ dst,
    const _Float16* __restrict__ gfrags,
    const float* __restrict__ b1, const float* __restrict__ b2, const float* __restrict__ b3,
    float* __restrict__ agg)
{
  extern __shared__ char smem_raw[];
  _Float16* smem = (_Float16*)smem_raw;

  // stage all MLP weight fragments (90112 B) into LDS
  {
    const uint4* gsrc = (const uint4*)gfrags;
    uint4* ldst = (uint4*)smem;
    for (int i = threadIdx.x; i < (LDS_ACT * 2) / 16; i += blockDim.x) ldst[i] = gsrc[i];
  }
  __syncthreads();

  const int lane = threadIdx.x & 31;
  const int warp = threadIdx.x >> 5;
  const int lo = lane & 15, hi = lane >> 4;
  const _Float16* w1f = smem;
  const _Float16* w2f = smem + LDS_W2F;
  const _Float16* w3f = smem + LDS_W3F;
  _Float16* h1s = smem + LDS_ACT + warp * 3072;   // 16 x 128 f16
  _Float16* h2s = h1s + 2048;                     // 16 x 64  f16

  const int numTiles = (EE + 15) >> 4;
  for (int tile = blockIdx.x * WAVES + warp; tile < numTiles; tile += gridDim.x * WAVES) {
    const int e0 = tile << 4;

    // ---- A matrix for layer 1: 16 edges x K(=3, padded to 32) ----
    v16h a1;
#pragma unroll
    for (int j = 0; j < 16; j++) a1[j] = (_Float16)0.f;
    if (lane < 16 && (e0 + lane) < EE) {
      const float* ea = edge_attr + (size_t)(e0 + lane) * 3;
      a1[0] = (_Float16)ea[0]; a1[1] = (_Float16)ea[1]; a1[2] = (_Float16)ea[2];
    }

    // ---- gather x[src] (C-layout: lane holds x[m=v+8*hi][lo]) and dst ids ----
    float xreg[8];
    int   dreg[8];
#pragma unroll
    for (int v = 0; v < 8; v++) {
      int e = e0 + v + hi * 8;
      if (e < EE) { xreg[v] = x[(size_t)src[e] * F_INC + lo]; dreg[v] = dst[e]; }
      else        { xreg[v] = 0.f; dreg[v] = -1; }
    }

    // ---- layer 1: 3 -> 128, relu, store f16 tile to LDS ----
#pragma unroll 2
    for (int t = 0; t < 8; t++) {
      v8f c = splat8(b1[t * 16 + lo]);
      c = wmma16(a1, load_frag16(w1f + ((size_t)t * 512 + lane * 16)), c);
#pragma unroll
      for (int v = 0; v < 8; v++)
        h1s[(v + hi * 8) * 128 + t * 16 + lo] = (_Float16)fmaxf(c[v], 0.f);
    }

    // ---- layer 2: 128 -> 64 ----
    v16h a2[4];
#pragma unroll
    for (int kg = 0; kg < 4; kg++) a2[kg] = load_act_frag(h1s, 128, kg, lo, hi);
#pragma unroll
    for (int t = 0; t < 4; t++) {
      v8f c = splat8(b2[t * 16 + lo]);
#pragma unroll
      for (int kg = 0; kg < 4; kg++)
        c = wmma16(a2[kg], load_frag16(w2f + ((size_t)(kg * 4 + t) * 512 + lane * 16)), c);
#pragma unroll
      for (int v = 0; v < 8; v++)
        h2s[(v + hi * 8) * 64 + t * 16 + lo] = (_Float16)fmaxf(c[v], 0.f);
    }

    // ---- layer 3: 64 -> 512, fused einsum msg[e][o] += x[src][i] * w_e[i][o] ----
    // Column block c = 16t + lo maps to (i = t>>1, o = 16*(t&1)+lo).  Process the
    // pair (t=2i, t=2i+1) together so each x broadcast (ds_bpermute) is used twice.
    v16h a3[2];
#pragma unroll
    for (int kg = 0; kg < 2; kg++) a3[kg] = load_act_frag(h2s, 64, kg, lo, hi);

    float macc0[8], macc1[8];   // o = lo  and  o = 16 + lo
#pragma unroll
    for (int v = 0; v < 8; v++) { macc0[v] = 0.f; macc1[v] = 0.f; }

#pragma unroll 4
    for (int i = 0; i < 16; i++) {
      const int t0 = 2 * i, t1 = 2 * i + 1;
      v8f c0 = splat8(b3[t0 * 16 + lo]);
      v8f c1 = splat8(b3[t1 * 16 + lo]);
#pragma unroll
      for (int kg = 0; kg < 2; kg++)
        c0 = wmma16(a3[kg], load_frag16(w3f + ((size_t)(kg * 32 + t0) * 512 + lane * 16)), c0);
#pragma unroll
      for (int kg = 0; kg < 2; kg++)
        c1 = wmma16(a3[kg], load_frag16(w3f + ((size_t)(kg * 32 + t1) * 512 + lane * 16)), c1);
#pragma unroll
      for (int v = 0; v < 8; v++) {
        float xv = __shfl(xreg[v], i + (hi << 4), 32);  // x[src[e0+v+8*hi]][i]
        macc0[v] += xv * c0[v];
        macc1[v] += xv * c1[v];
      }
    }

    // ---- scatter msg into agg[dst][0..31] ----
#pragma unroll
    for (int v = 0; v < 8; v++) {
      if (dreg[v] >= 0) {
        float* p = agg + (size_t)dreg[v] * 32 + lo;
        atomicAdd(p,      macc0[v]);
        atomicAdd(p + 16, macc1[v]);
      }
    }
  }
}

// ---------------------------------------------------------------------------
// Kernel 2: h1 = relu(x @ root_w + agg + b), in place over agg [N,32]
// ---------------------------------------------------------------------------
__global__ void root_combine_kernel(const float* __restrict__ x, const float* __restrict__ rw,
                                    const float* __restrict__ b, float* __restrict__ agg) {
  int tid = blockIdx.x * blockDim.x + threadIdx.x;
  if (tid >= NN * 32) return;
  int n = tid >> 5, o = tid & 31;
  float s = b[o] + agg[tid];
  const float* xr = x + (size_t)n * F_INC;
#pragma unroll
  for (int i = 0; i < 16; i++) s += xr[i] * rw[i * 32 + o];
  agg[tid] = fmaxf(s, 0.f);
}

// ---------------------------------------------------------------------------
// Kernel 3: WMMA GEMM  C[N,64] = A[N,K] @ W   (K = 32*KG, no bias)
// ---------------------------------------------------------------------------
template <int KG>
__global__ __launch_bounds__(128) void gat_gemm_kernel(
    const float* __restrict__ A, const _Float16* __restrict__ wfrag, float* __restrict__ C, int K)
{
  const int lane = threadIdx.x & 31;
  const int warp = threadIdx.x >> 5;
  const int lo = lane & 15, hi = lane >> 4;
  const int numTiles = (NN + 15) >> 4;
  for (int tile = blockIdx.x * 4 + warp; tile < numTiles; tile += gridDim.x * 4) {
    const int n0 = tile << 4;
    const int row = n0 + lo;
    v16h a[KG];
#pragma unroll
    for (int kg = 0; kg < KG; kg++) {
      v16h af;
#pragma unroll
      for (int j = 0; j < 16; j++) {
        int k = kg * 32 + ((j >> 3) << 4) + hi * 8 + (j & 7);
        float v = 0.f;
        if (row < NN && k < K) v = A[(size_t)row * K + k];
        af[j] = (_Float16)v;
      }
      a[kg] = af;
    }
#pragma unroll
    for (int t = 0; t < 4; t++) {
      v8f c = splat8(0.f);
#pragma unroll
      for (int kg = 0; kg < KG; kg++)
        c = wmma16(a[kg], load_frag16(wfrag + ((size_t)(kg * 4 + t) * 512 + lane * 16)), c);
#pragma unroll
      for (int v = 0; v < 8; v++) {
        int m = n0 + v + 8 * hi;
        if (m < NN) C[(size_t)m * 64 + t * 16 + lo] = c[v];
      }
    }
  }
}

// ---------------------------------------------------------------------------
// GAT attention pipeline kernels
// ---------------------------------------------------------------------------
__global__ void node_dots_kernel(const float* __restrict__ h, const float* __restrict__ av1,
                                 const float* __restrict__ av2, float* __restrict__ o1,
                                 float* __restrict__ o2) {
  int n = blockIdx.x * blockDim.x + threadIdx.x;
  if (n >= NN) return;
  const float* hr = h + (size_t)n * 64;
  float s1 = 0.f, s2 = 0.f;
#pragma unroll 8
  for (int c = 0; c < 64; c++) { s1 += hr[c] * av1[c]; s2 += hr[c] * av2[c]; }
  o1[n] = s1; o2[n] = s2;
}

__global__ void fill_f32_kernel(float* __restrict__ p, float v, int n) {
  int i = blockIdx.x * blockDim.x + threadIdx.x;
  if (i < n) p[i] = v;
}

__global__ void edge_max_kernel(const int* __restrict__ src, const int* __restrict__ dst,
                                const float* __restrict__ as, const float* __restrict__ ad,
                                float* __restrict__ m) {
  int e = blockIdx.x * blockDim.x + threadIdx.x;
  if (e >= EE) return;
  int d = dst[e];
  atomicMaxF(&m[d], leaky(as[src[e]] + ad[d]));
}

__global__ void self_max_kernel(const float* __restrict__ as, const float* __restrict__ ad,
                                float* __restrict__ m) {
  int n = blockIdx.x * blockDim.x + threadIdx.x;
  if (n >= NN) return;
  m[n] = fmaxf(m[n], leaky(as[n] + ad[n]));
}

__global__ void edge_expagg_kernel(const int* __restrict__ src, const int* __restrict__ dst,
                                   const float* __restrict__ as, const float* __restrict__ ad,
                                   const float* __restrict__ m, float* __restrict__ s,
                                   const float* __restrict__ hlin, float* __restrict__ hacc) {
  long long tid = (long long)blockIdx.x * blockDim.x + threadIdx.x;
  if (tid >= (long long)EE * 64) return;
  int e = (int)(tid >> 6), c = (int)(tid & 63);
  int sN = src[e], dN = dst[e];
  float ee = __expf(leaky(as[sN] + ad[dN]) - m[dN]);
  if (c == 0) atomicAdd(&s[dN], ee);
  atomicAdd(&hacc[(size_t)dN * 64 + c], ee * hlin[(size_t)sN * 64 + c]);
}

__global__ void self_expagg_kernel(const float* __restrict__ as, const float* __restrict__ ad,
                                   const float* __restrict__ m, float* __restrict__ s,
                                   const float* __restrict__ hlin, float* __restrict__ hacc) {
  int tid = blockIdx.x * blockDim.x + threadIdx.x;
  if (tid >= NN * 64) return;
  int n = tid >> 6, c = tid & 63;
  float ee = __expf(leaky(as[n] + ad[n]) - m[n]);
  if (c == 0) s[n] += ee;
  hacc[tid] += ee * hlin[tid];
}

__global__ void gat_final_kernel(float* __restrict__ hacc, const float* __restrict__ s,
                                 const float* __restrict__ b) {
  int tid = blockIdx.x * blockDim.x + threadIdx.x;
  if (tid >= NN * 64) return;
  hacc[tid] = fmaxf(hacc[tid] / s[tid >> 6] + b[tid & 63], 0.f);
}

// ---------------------------------------------------------------------------
// global mean pool
// ---------------------------------------------------------------------------
__global__ void pool_acc_kernel(const float* __restrict__ h3, const int* __restrict__ batch,
                                float* __restrict__ pool, float* __restrict__ cnt) {
  int tid = blockIdx.x * blockDim.x + threadIdx.x;
  if (tid >= NN * 64) return;
  int n = tid >> 6, c = tid & 63;
  int g = batch[n];
  atomicAdd(&pool[g * 64 + c], h3[tid]);
  if (c == 0) atomicAdd(&cnt[g], 1.f);
}

__global__ void pool_final_kernel(const float* __restrict__ pool, const float* __restrict__ cnt,
                                  float* __restrict__ out) {
  int tid = blockIdx.x * blockDim.x + threadIdx.x;
  if (tid >= 32 * 64) return;
  out[tid] = pool[tid] / fmaxf(cnt[tid >> 6], 1.f);
}

// ---------------------------------------------------------------------------
// host driver
// ---------------------------------------------------------------------------
extern "C" void kernel_launch(void* const* d_in, const int* in_sizes, int n_in,
                              void* d_out, int out_size, void* d_ws, size_t ws_size,
                              hipStream_t stream) {
  (void)in_sizes; (void)n_in; (void)out_size; (void)ws_size;
  const float* x     = (const float*)d_in[0];
  const float* eatt  = (const float*)d_in[1];
  const int*   eidx  = (const int*)d_in[2];
  const int*   batch = (const int*)d_in[3];
  const float* w1 = (const float*)d_in[4];  const float* b1 = (const float*)d_in[5];
  const float* w2 = (const float*)d_in[6];  const float* b2 = (const float*)d_in[7];
  const float* w3 = (const float*)d_in[8];  const float* b3 = (const float*)d_in[9];
  const float* rootw = (const float*)d_in[10]; const float* cb = (const float*)d_in[11];
  const float* g1w = (const float*)d_in[12]; const float* g1s = (const float*)d_in[13];
  const float* g1d = (const float*)d_in[14]; const float* g1b = (const float*)d_in[15];
  const float* g2w = (const float*)d_in[16]; const float* g2s = (const float*)d_in[17];
  const float* g2d = (const float*)d_in[18]; const float* g2b = (const float*)d_in[19];
  const int* srcp = eidx;
  const int* dstp = eidx + EE;
  float* out = (float*)d_out;

  char* ws = (char*)d_ws;
  _Float16* fragmlp = (_Float16*)(ws);                 // 90112 B (w1f|w2f|w3f contiguous)
  _Float16* fragg1  = (_Float16*)(ws + 90112);         // 4096 B
  _Float16* fragg2  = (_Float16*)(ws + 94208);         // 8192 B
  float* agg   = (float*)(ws + 102400);                                  // N*32
  float* hlin  = (float*)(ws + 102400 + 6400000LL);                      // N*64
  float* h2    = (float*)(ws + 102400 + 6400000LL + 12800000LL);         // N*64
  float* h3    = (float*)(ws + 102400 + 6400000LL + 2 * 12800000LL);     // N*64
  float* asrcv = (float*)(ws + 44902400LL);
  float* adstv = (float*)(ws + 45102400LL);
  float* mnode = (float*)(ws + 45302400LL);
  float* snode = (float*)(ws + 45502400LL);
  float* pool  = (float*)(ws + 45702400LL);
  float* cnt   = (float*)(ws + 45710592LL);

  // --- pack weights into WMMA B-fragments ---
  prep_frags_kernel<<<16,  256, 0, stream>>>(w1,  fragmlp,          3,  128, 1);
  prep_frags_kernel<<<32,  256, 0, stream>>>(w2,  fragmlp + 4096,   128, 64, 4);
  prep_frags_kernel<<<128, 256, 0, stream>>>(w3,  fragmlp + 12288,  64, 512, 2);
  prep_frags_kernel<<<8,   256, 0, stream>>>(g1w, fragg1,           32,  64, 1);
  prep_frags_kernel<<<16,  256, 0, stream>>>(g2w, fragg2,           64,  64, 2);

  // --- fused EdgeMLP + NNConv message + scatter ---
  hipMemsetAsync(agg, 0, (size_t)NN * 32 * sizeof(float), stream);
  nnconv_edge_kernel<<<384, WAVES * 32, LDS_HALVES * 2, stream>>>(
      x, eatt, srcp, dstp, fragmlp, b1, b2, b3, agg);

  // --- root transform + bias + relu -> h1 (in place over agg) ---
  root_combine_kernel<<<(NN * 32 + 255) / 256, 256, 0, stream>>>(x, rootw, cb, agg);

  // --- GAT layer runner ---
  auto run_gat = [&](const float* hin, int K, int KG, const _Float16* wf,
                     const float* avs, const float* avd, const float* gb, float* hout) {
    if (KG == 1) gat_gemm_kernel<1><<<512, 128, 0, stream>>>(hin, wf, hlin, K);
    else         gat_gemm_kernel<2><<<512, 128, 0, stream>>>(hin, wf, hlin, K);
    node_dots_kernel<<<(NN + 255) / 256, 256, 0, stream>>>(hlin, avs, avd, asrcv, adstv);
    fill_f32_kernel<<<(NN + 255) / 256, 256, 0, stream>>>(mnode, -3.0e38f, NN);
    hipMemsetAsync(snode, 0, (size_t)NN * sizeof(float), stream);
    hipMemsetAsync(hout, 0, (size_t)NN * 64 * sizeof(float), stream);
    edge_max_kernel<<<(EE + 255) / 256, 256, 0, stream>>>(srcp, dstp, asrcv, adstv, mnode);
    self_max_kernel<<<(NN + 255) / 256, 256, 0, stream>>>(asrcv, adstv, mnode);
    long long et = (long long)EE * 64;
    edge_expagg_kernel<<<(unsigned)((et + 255) / 256), 256, 0, stream>>>(
        srcp, dstp, asrcv, adstv, mnode, snode, hlin, hout);
    self_expagg_kernel<<<(NN * 64 + 255) / 256, 256, 0, stream>>>(
        asrcv, adstv, mnode, snode, hlin, hout);
    gat_final_kernel<<<(NN * 64 + 255) / 256, 256, 0, stream>>>(hout, snode, gb);
  };

  run_gat(agg, 32, 1, fragg1, g1s, g1d, g1b, h2);   // h1 -> h2
  run_gat(h2,  64, 2, fragg2, g2s, g2d, g2b, h3);   // h2 -> h3

  // --- global mean pool ---
  hipMemsetAsync(pool, 0, 32 * 64 * sizeof(float), stream);
  hipMemsetAsync(cnt, 0, 32 * sizeof(float), stream);
  pool_acc_kernel<<<(NN * 64 + 255) / 256, 256, 0, stream>>>(h3, batch, pool, cnt);
  pool_final_kernel<<<(32 * 64 + 255) / 256, 256, 0, stream>>>(pool, cnt, out);
}
